// GATEncoder_64012192580152
// MI455X (gfx1250) — matrix operand order repaired and checked by reading, not compiled, not run on hardware
//
#include <hip/hip_runtime.h>
#include <hip/hip_bf16.h>

// ---------------------------------------------------------------------------
// GAT encoder for MI455X (gfx1250): WMMA f16 GEMMs + L2-resident edge phase.
// ---------------------------------------------------------------------------

typedef __attribute__((ext_vector_type(16))) _Float16 v16h;
typedef __attribute__((ext_vector_type(8)))  _Float16 v8h;
typedef __attribute__((ext_vector_type(8)))  float    v8f;

#define NEG_SLOPE 0.2f
#define LN_EPS    1e-5f

// ---------------- order-preserving float <-> uint for atomic max -----------
__device__ __forceinline__ unsigned enc_f32(float f) {
  unsigned u = __float_as_uint(f);
  return (u & 0x80000000u) ? ~u : (u | 0x80000000u);
}
__device__ __forceinline__ float dec_f32(unsigned u) {
  return __uint_as_float((u & 0x80000000u) ? (u & 0x7FFFFFFFu) : ~u);
}

// ---------------- f32 -> f16 conversion ------------------------------------
__global__ void cvt_f16_kernel(const float* __restrict__ in,
                               _Float16* __restrict__ out, int n) {
  int i = blockIdx.x * blockDim.x + threadIdx.x;
  if (i < n) out[i] = (_Float16)in[i];
}

// W[K][N] -> Wt[N][K] in f16 (so WMMA B-fragment reads are contiguous in K)
__global__ void transpose_f16_kernel(const float* __restrict__ W,
                                     _Float16* __restrict__ Wt, int K, int N) {
  int i = blockIdx.x * blockDim.x + threadIdx.x;
  if (i >= K * N) return;
  int n = i / K, k = i % K;
  Wt[i] = (_Float16)W[(size_t)k * N + n];
}

// ---------------- WMMA GEMM: C[MxN] = A[MxK] * Bt[NxK]^T -------------------
// One wave per 16x16 C tile, K stepped by 32 through v_wmma_f32_16x16x32_f16.
// Fragment layouts per CDNA5 ISA 7.12.2:
//  A (16x32 f16): lanes 0-15 row M=lane, K in {0..7,16..23};
//                 lanes 16-31 row M=lane-16, K in {8..15,24..31}.
//  B (32x16 f16): lane holds column N=lane&15; lanes 0-15 K in {0..7,16..23},
//                 lanes 16-31 K in {8..15,24..31} (same as A, from Bt row).
//  C (16x16 f32): VGPR r -> row r (lanes 0-15) / r+8 (lanes 16-31), col=lane&15.
__global__ __launch_bounds__(128) void gemm_f16_wmma_kernel(
    const _Float16* __restrict__ A, const _Float16* __restrict__ Bt,
    float* __restrict__ C, int M, int N, int K) {
  const int ntiles = N >> 4;
  const int total  = (M >> 4) * ntiles;
  const int wave   = blockIdx.x * (blockDim.x >> 5) + (threadIdx.x >> 5);
  if (wave >= total) return;           // wave-uniform: EXEC all-1s inside
  const int lane = threadIdx.x & 31;
  const int tm = (wave / ntiles) << 4;
  const int tn = (wave % ntiles) << 4;
  const int hi = lane >> 4;            // lane-group 0/1
  const int lr = lane & 15;

  const _Float16* arow = A  + (size_t)(tm + lr) * K;
  const _Float16* brow = Bt + (size_t)(tn + lr) * K;

  v8f acc = {};
  for (int k = 0; k < K; k += 32) {
    union { v16h v; v8h h[2]; } a, b;
    a.h[0] = *(const v8h*)(arow + k + (hi << 3));
    a.h[1] = *(const v8h*)(arow + k + 16 + (hi << 3));
    b.h[0] = *(const v8h*)(brow + k + (hi << 3));
    b.h[1] = *(const v8h*)(brow + k + 16 + (hi << 3));
    acc = __builtin_amdgcn_wmma_f32_16x16x32_f16(
        false, a.v, false, b.v, (short)0, acc, false, false);
  }
  float* crow = C + (size_t)(tm + hi * 8) * N + tn + lr;
#pragma unroll
  for (int r = 0; r < 8; ++r) crow[(size_t)r * N] = acc[r];
}

// ---------------- per-node attention scores (one wave per node) ------------
__global__ __launch_bounds__(128) void node_scores_kernel(
    const float* __restrict__ h, const float* __restrict__ att_s,
    const float* __restrict__ att_d, float* __restrict__ as_out,
    float* __restrict__ ad_out, int n_nodes, int heads, int C) {
  int node = blockIdx.x * (blockDim.x >> 5) + (threadIdx.x >> 5);
  if (node >= n_nodes) return;
  int lane = threadIdx.x & 31;
  const float* row = h + (size_t)node * heads * C;
  for (int hh = 0; hh < heads; ++hh) {
    float ps = 0.f, pd = 0.f;
    for (int c = lane; c < C; c += 32) {
      float v = row[hh * C + c];
      ps += v * att_s[hh * C + c];
      pd += v * att_d[hh * C + c];
    }
    for (int off = 16; off > 0; off >>= 1) {
      ps += __shfl_xor(ps, off);
      pd += __shfl_xor(pd, off);
    }
    if (lane == 0) {
      as_out[node * heads + hh] = ps;
      ad_out[node * heads + hh] = pd;
    }
  }
}

// ---------------- edge score + segment max (thread per edge) ---------------
__global__ void edge_scores_kernel(const int* __restrict__ src,
                                   const int* __restrict__ dst,
                                   const float* __restrict__ as_in,
                                   const float* __restrict__ ad_in,
                                   float* __restrict__ ebuf,
                                   unsigned* __restrict__ mx,
                                   int E_real, int E_total, int heads) {
  int e = blockIdx.x * blockDim.x + threadIdx.x;
  if (e >= E_total) return;
  int s = (e < E_real) ? src[e] : (e - E_real);
  int d = (e < E_real) ? dst[e] : (e - E_real);
  for (int hh = 0; hh < heads; ++hh) {
    float sc = as_in[s * heads + hh] + ad_in[d * heads + hh];
    sc = sc > 0.f ? sc : NEG_SLOPE * sc;          // leaky relu
    ebuf[(size_t)e * heads + hh] = sc;
    atomicMax(&mx[d * heads + hh], enc_f32(sc));
  }
}

// ---------------- exp(e - max) + segment sum -------------------------------
__global__ void edge_exp_kernel(const int* __restrict__ src,
                                const int* __restrict__ dst,
                                float* __restrict__ ebuf,
                                const unsigned* __restrict__ mx,
                                float* __restrict__ denom,
                                int E_real, int E_total, int heads) {
  int e = blockIdx.x * blockDim.x + threadIdx.x;
  if (e >= E_total) return;
  int d = (e < E_real) ? dst[e] : (e - E_real);
  for (int hh = 0; hh < heads; ++hh) {
    float ex = expf(ebuf[(size_t)e * heads + hh] - dec_f32(mx[d * heads + hh]));
    ebuf[(size_t)e * heads + hh] = ex;
    atomicAdd(&denom[d * heads + hh], ex);
  }
}

// ---------------- weighted message scatter-add (block per edge) ------------
// blockDim == heads*C; contiguous atomic adds per edge -> coalesced, L2-hit.
__global__ void aggregate_kernel(const int* __restrict__ src,
                                 const int* __restrict__ dst,
                                 const float* __restrict__ ebuf,
                                 const float* __restrict__ denom,
                                 const float* __restrict__ hfeat,
                                 float* __restrict__ out,
                                 int E_real, int heads, int C) {
  int e = blockIdx.x;
  int t = threadIdx.x;
  int s = (e < E_real) ? src[e] : (e - E_real);
  int d = (e < E_real) ? dst[e] : (e - E_real);
  int hh = t / C;
  float alpha = ebuf[(size_t)e * heads + hh] / (denom[d * heads + hh] + 1e-16f);
  int F = heads * C;
  atomicAdd(&out[(size_t)d * F + t], alpha * hfeat[(size_t)s * F + t]);
}

// ---------------- bias + ELU + convert to f16 (layer-2 GEMM input) ---------
__global__ void bias_elu_f16_kernel(const float* __restrict__ in,
                                    const float* __restrict__ bias,
                                    _Float16* __restrict__ out,
                                    int total, int F) {
  int i = blockIdx.x * blockDim.x + threadIdx.x;
  if (i >= total) return;
  float v = in[i] + bias[i % F];
  v = v > 0.f ? v : (expf(v) - 1.f);               // ELU
  out[i] = (_Float16)v;
}

// ---------------- fused bias + LayerNorm (block per node, C==128) ----------
__global__ __launch_bounds__(128) void bias_layernorm_kernel(
    const float* __restrict__ in, const float* __restrict__ bias,
    const float* __restrict__ gamma, const float* __restrict__ beta,
    float* __restrict__ out, int C) {
  int n = blockIdx.x, t = threadIdx.x;
  float v = in[(size_t)n * C + t] + bias[t];
  float s = v, q = v * v;
  for (int off = 16; off > 0; off >>= 1) {
    s += __shfl_xor(s, off);
    q += __shfl_xor(q, off);
  }
  __shared__ float ss[4], sq[4];
  int w = t >> 5;
  if ((t & 31) == 0) { ss[w] = s; sq[w] = q; }
  __syncthreads();
  float S = ss[0] + ss[1] + ss[2] + ss[3];
  float Q = sq[0] + sq[1] + sq[2] + sq[3];
  float mu  = S / C;
  float var = Q / C - mu * mu;
  out[(size_t)n * C + t] = (v - mu) * rsqrtf(var + LN_EPS) * gamma[t] + beta[t];
}

// ---------------------------------------------------------------------------
static inline size_t carve(size_t& off, size_t bytes) {
  size_t p = off;
  off = (off + bytes + 255) & ~(size_t)255;
  return p;
}
static inline int cdiv(long long a, long long b) { return (int)((a + b - 1) / b); }

extern "C" void kernel_launch(void* const* d_in, const int* in_sizes, int n_in,
                              void* d_out, int out_size, void* d_ws, size_t ws_size,
                              hipStream_t stream) {
  const int IN_C = 128, HID = 64, HEADS = 4, OUT_C = 128;
  const int F1 = HEADS * HID;                      // 256
  const int N  = in_sizes[0] / IN_C;               // 50000 nodes
  const int E  = in_sizes[1] / 2;                  // 800000 edges
  const int ET = E + N;                            // + self loops

  const float* x        = (const float*)d_in[0];
  const int*   src      = (const int*)d_in[1];
  const int*   dst      = src + E;
  const float* W1       = (const float*)d_in[2];
  const float* att_src1 = (const float*)d_in[3];
  const float* att_dst1 = (const float*)d_in[4];
  const float* b1       = (const float*)d_in[5];
  const float* W2       = (const float*)d_in[6];
  const float* att_src2 = (const float*)d_in[7];
  const float* att_dst2 = (const float*)d_in[8];
  const float* b2       = (const float*)d_in[9];
  const float* gamma    = (const float*)d_in[10];
  const float* beta     = (const float*)d_in[11];

  // ---- workspace carve (aliased: h2<-h1 region, out2<-out1, e2<-e1) ----
  char* ws = (char*)d_ws;
  size_t off = 0;
  float*    h1   = (float*)   (ws + carve(off, (size_t)N * F1 * 4));   // 51.2MB
  float*    o1   = (float*)   (ws + carve(off, (size_t)N * F1 * 4));   // 51.2MB
  _Float16* xh   = (_Float16*)(ws + carve(off, (size_t)N * IN_C * 2)); // 12.8MB
  _Float16* h1h  = (_Float16*)(ws + carve(off, (size_t)N * F1 * 2));   // 25.6MB
  _Float16* w1t  = (_Float16*)(ws + carve(off, (size_t)F1 * IN_C * 2));
  _Float16* w2t  = (_Float16*)(ws + carve(off, (size_t)OUT_C * F1 * 2));
  float*    e1   = (float*)   (ws + carve(off, (size_t)ET * HEADS * 4)); // 13.6MB
  float*    as1  = (float*)   (ws + carve(off, (size_t)N * HEADS * 4));
  float*    ad1  = (float*)   (ws + carve(off, (size_t)N * HEADS * 4));
  unsigned* m1   = (unsigned*)(ws + carve(off, (size_t)N * HEADS * 4));
  float*    dn1  = (float*)   (ws + carve(off, (size_t)N * HEADS * 4));
  float*    as2  = (float*)   (ws + carve(off, (size_t)N * 4));
  float*    ad2  = (float*)   (ws + carve(off, (size_t)N * 4));
  unsigned* m2   = (unsigned*)(ws + carve(off, (size_t)N * 4));
  float*    dn2  = (float*)   (ws + carve(off, (size_t)N * 4));
  float*    h2   = h1;   // alias: h1 dead after aggregate1
  float*    o2   = o1;   // alias: o1 dead after bias_elu
  float*    e2   = e1;   // alias: e1 dead after aggregate1

  // ---- precision prep: f16 copies for WMMA ----
  cvt_f16_kernel<<<cdiv((long long)N * IN_C, 256), 256, 0, stream>>>(x, xh, N * IN_C);
  transpose_f16_kernel<<<cdiv(IN_C * F1, 256), 256, 0, stream>>>(W1, w1t, IN_C, F1);
  transpose_f16_kernel<<<cdiv(F1 * OUT_C, 256), 256, 0, stream>>>(W2, w2t, F1, OUT_C);

  // ---- zero accumulators ----
  hipMemsetAsync(o1, 0, (size_t)N * F1 * 4, stream);
  hipMemsetAsync(m1, 0, (size_t)N * HEADS * 4, stream);
  hipMemsetAsync(dn1, 0, (size_t)N * HEADS * 4, stream);

  // ---- layer 1 ----
  {
    int tiles = (N / 16) * (F1 / 16);
    gemm_f16_wmma_kernel<<<cdiv(tiles, 4), 128, 0, stream>>>(xh, w1t, h1, N, F1, IN_C);
  }
  node_scores_kernel<<<cdiv(N, 4), 128, 0, stream>>>(h1, att_src1, att_dst1,
                                                     as1, ad1, N, HEADS, HID);
  edge_scores_kernel<<<cdiv(ET, 256), 256, 0, stream>>>(src, dst, as1, ad1, e1, m1,
                                                        E, ET, HEADS);
  edge_exp_kernel<<<cdiv(ET, 256), 256, 0, stream>>>(src, dst, e1, m1, dn1,
                                                     E, ET, HEADS);
  aggregate_kernel<<<ET, F1, 0, stream>>>(src, dst, e1, dn1, h1, o1, E, HEADS, HID);
  bias_elu_f16_kernel<<<cdiv((long long)N * F1, 256), 256, 0, stream>>>(
      o1, b1, h1h, N * F1, F1);

  // ---- layer 2 (aliased buffers re-zeroed mid-graph) ----
  hipMemsetAsync(o2, 0, (size_t)N * OUT_C * 4, stream);
  hipMemsetAsync(m2, 0, (size_t)N * 4, stream);
  hipMemsetAsync(dn2, 0, (size_t)N * 4, stream);
  {
    int tiles = (N / 16) * (OUT_C / 16);
    gemm_f16_wmma_kernel<<<cdiv(tiles, 4), 128, 0, stream>>>(h1h, w2t, h2, N, OUT_C, F1);
  }
  node_scores_kernel<<<cdiv(N, 4), 128, 0, stream>>>(h2, att_src2, att_dst2,
                                                     as2, ad2, N, 1, OUT_C);
  edge_scores_kernel<<<cdiv(ET, 256), 256, 0, stream>>>(src, dst, as2, ad2, e2, m2,
                                                        E, ET, 1);
  edge_exp_kernel<<<cdiv(ET, 256), 256, 0, stream>>>(src, dst, e2, m2, dn2,
                                                     E, ET, 1);
  aggregate_kernel<<<ET, OUT_C, 0, stream>>>(src, dst, e2, dn2, h2, o2, E, 1, OUT_C);

  // ---- fused bias + LayerNorm -> d_out ----
  bias_layernorm_kernel<<<N, OUT_C, 0, stream>>>(o2, b2, gamma, beta,
                                                 (float*)d_out, OUT_C);
}
